// RobustSupplyChainSAGE_2336462209551
// MI455X (gfx1250) — matrix operand order repaired and checked by reading, not compiled, not run on hardware
//
#include <hip/hip_runtime.h>
#include <math.h>

typedef __attribute__((ext_vector_type(16))) _Float16 v16h;
typedef __attribute__((ext_vector_type(8)))  float    v8f;
typedef __attribute__((ext_vector_type(4)))  unsigned v4u;
typedef __attribute__((ext_vector_type(8)))  unsigned v8u;

#define LN_EPS 1e-5f
#define BN_EPS 1e-5f
#define ENC_NEG_INF 0x007FFFFFu  // order-preserving encoding of -inf

__device__ __forceinline__ unsigned encf(float f) {
  unsigned b = __float_as_uint(f);
  return (b & 0x80000000u) ? ~b : (b | 0x80000000u);
}
__device__ __forceinline__ float decf(unsigned u) {
  unsigned b = (u & 0x80000000u) ? (u & 0x7fffffffu) : ~u;
  return __uint_as_float(b);
}
__device__ __forceinline__ float gelu_exact(float x) {
  return 0.5f * x * (1.0f + erff(x * 0.7071067811865475f));
}

// ---- A-fragment loader (CDNA5 wave32 16x32 f16 layout, CDNA5 ISA §7.12.2) ----
template <typename F>
__device__ __forceinline__ v16h load_a_frag(int lane, F fetch) {
  const int kh = lane >> 4;
  v16h a;
#pragma unroll
  for (int v = 0; v < 8; ++v) {
    int k = (v < 4) ? (8 * kh + 2 * v) : (16 + 8 * kh + 2 * (v - 4));
    a[2 * v]     = (_Float16)fetch(k);
    a[2 * v + 1] = (_Float16)fetch(k + 1);
  }
  return a;
}

// Pre-swizzled B fragment: one 32B-per-lane coalesced load.
__device__ __forceinline__ v16h load_b_packed(const _Float16* __restrict__ base,
                                              int frag, int lane) {
  return ((const v16h*)base)[frag * 32 + lane];
}

#define WMMA(a, b, c) \
  __builtin_amdgcn_wmma_f32_16x16x32_f16(false, (a), false, (b), (short)0, (c), false, false)

// ---- CDNA5 async global->LDS copy (ASYNCcnt path, CDNA5 ISA) ----
__device__ __forceinline__ void async_b128_to_lds(unsigned lds_addr, unsigned gofs,
                                                  const float* base) {
  asm volatile("global_load_async_to_lds_b128 %0, %1, %2"
               :: "v"(lds_addr), "v"(gofs), "s"(base)
               : "memory");
}
__device__ __forceinline__ void wait_asynccnt0() {
  asm volatile("s_wait_asynccnt 0x0" ::: "memory");
}

// ---- CDNA5 TDM gather: 8 rows of 64 f32, 32-bit indices, pad-interleaved into
// a [rows][128] f32 LDS tile (CDNA5 ISA-8). All operands wave-uniform.
__device__ __forceinline__ void tdm_gather8(const float* base, unsigned lds_addr,
                                            unsigned n_rows,
                                            unsigned i0, unsigned i1, unsigned i2, unsigned i3,
                                            unsigned i4, unsigned i5, unsigned i6, unsigned i7) {
  unsigned long long ga = (unsigned long long)(uintptr_t)base;
  v4u g0 = { 0xC0000001u,                         // count=1, idx32, gather_mode
             lds_addr,
             (unsigned)ga,
             (unsigned)(ga >> 32) | 0x80000000u };// type=2
  v8u g1 = { 0x7F520000u,                         // 4B elems, pad 64dw every 64dw
             64u << 16,                           // tensor_dim0 = 64
             (n_rows & 0xFFFFu) << 16,            // tensor_dim1 lo
             (n_rows >> 16) | (64u << 16),        // tensor_dim1 hi | tile_dim0=64
             8u,                                  // tile_dim1 = #indices
             64u,                                 // tensor_dim0_stride = 64
             0u, 0u };
  v4u g2 = { i0, i1, i2, i3 };
  v4u g3 = { i4, i5, i6, i7 };
  asm volatile("tensor_load_to_lds %0, %1, %2, %3"
               :: "s"(g0), "s"(g1), "s"(g2), "s"(g3)
               : "memory");
}

// ---------------- utility kernels ----------------
__global__ void fill_f32(float* __restrict__ p, float v, int n) {
  int i = blockIdx.x * blockDim.x + threadIdx.x;
  if (i < n) p[i] = v;
}
__global__ void fill_u32(unsigned* __restrict__ p, unsigned v, int n) {
  int i = blockIdx.x * blockDim.x + threadIdx.x;
  if (i < n) p[i] = v;
}
__global__ void cvt_f16(const float* __restrict__ src, _Float16* __restrict__ dst, int n) {
  int i = blockIdx.x * blockDim.x + threadIdx.x;
  if (i < n) dst[i] = (_Float16)src[i];
}
__global__ void deg_accum(const int* __restrict__ ei, int E, float* __restrict__ deg) {
  int e = blockIdx.x * blockDim.x + threadIdx.x;
  if (e < E) atomicAdd(&deg[ei[E + e]], 1.0f);
}

// Pre-swizzle a (possibly concatenated, zero-padded) f32 weight matrix into the
// wave32 B-fragment layout: dst[frag*512 + lane*16 + e] = W[ks*32 + (e<8 ? 8*(lane>>4)+e : 16+8*(lane>>4)+e-8)][nt*16 + lane&15]
__global__ void prep_frags(const float* __restrict__ srcA, const float* __restrict__ srcB,
                           int Ksplit, int Ktot, int Nn, _Float16* __restrict__ dst,
                           int total) {
  int i = blockIdx.x * blockDim.x + threadIdx.x;
  if (i >= total) return;
  int e    = i & 15;
  int lane = (i >> 4) & 31;
  int frag = i >> 9;
  int NT = Nn >> 4;
  int nt = frag % NT, ks = frag / NT;
  int gk  = ks * 32 + ((e < 8) ? (8 * (lane >> 4) + e) : (16 + 8 * (lane >> 4) + (e - 8)));
  int col = nt * 16 + (lane & 15);
  float v = 0.0f;
  if (gk < Ksplit)    v = srcA[gk * Nn + col];
  else if (gk < Ktot) v = srcB[(gk - Ksplit) * Nn + col];
  dst[i] = (_Float16)v;
}

// ---------------- node encoder: h = x @ W_enc + b_enc ----------------
__global__ __launch_bounds__(256) void encode_kernel(
    const _Float16* __restrict__ xh, const _Float16* __restrict__ fWenc,
    const float* __restrict__ benc, float* __restrict__ h, int N) {
  const int lane = threadIdx.x & 31, wid = threadIdx.x >> 5;
  const int row0 = blockIdx.x * 128 + wid * 16;
  const int m = lane & 15, kh = lane >> 4, n = m;
  const int arow = min(row0 + m, N - 1);

  union { v16h v; uint4 q[2]; } au;
  const _Float16* xr = xh + (size_t)arow * 32;
  au.q[0] = *(const uint4*)(xr + 8 * kh);
  au.q[1] = *(const uint4*)(xr + 16 + 8 * kh);
  v16h a = au.v;

  v8f acc[4] = {};
#pragma unroll
  for (int nt = 0; nt < 4; ++nt)
    acc[nt] = WMMA(a, load_b_packed(fWenc, nt, lane), acc[nt]);

#pragma unroll
  for (int r = 0; r < 8; ++r) {
    int row = row0 + 8 * kh + r;
    if (row < N) {
#pragma unroll
      for (int nt = 0; nt < 4; ++nt) {
        int col = nt * 16 + n;
        h[(size_t)row * 64 + col] = acc[nt][r] + benc[col];
      }
    }
  }
}

// ---------------- edge scatter: one wave per edge, 2 channels/lane ----------------
__global__ __launch_bounds__(256) void scatter_kernel(
    const int* __restrict__ ei, int E, const float* __restrict__ h,
    float* __restrict__ sum, unsigned* __restrict__ mx) {
  const int lane = threadIdx.x & 31;
  const int e = blockIdx.x * 8 + (threadIdx.x >> 5);
  if (e >= E) return;
  if (lane == 0) {                       // lookahead prefetch of a future source row
    int pe = e + 4096;
    if (pe < E) __builtin_prefetch(h + (size_t)ei[pe] * 64, 0, 0);
  }
  const int s = ei[e], d = ei[E + e];
  const float2 v = *(const float2*)(h + (size_t)s * 64 + lane * 2);
  float* sp = sum + (size_t)d * 64 + lane * 2;
  atomicAdd(sp, v.x);
  atomicAdd(sp + 1, v.y);
  unsigned* mp = mx + (size_t)d * 64 + lane * 2;
  atomicMax(mp, encf(v.x));
  atomicMax(mp + 1, encf(v.y));
}

// ---- SAGE combine: h = residual + GELU(LN([mean|max|h] @ [Wl;Wr] + bl)) ----
__global__ __launch_bounds__(256) void combine_kernel(
    float* __restrict__ h, const float* __restrict__ sum,
    const unsigned* __restrict__ mx, const float* __restrict__ deg,
    const _Float16* __restrict__ fWlr, const float* __restrict__ bl,
    const float* __restrict__ lng, const float* __restrict__ lnb, int N) {
  __shared__ float tile[128][65];
  const int lane = threadIdx.x & 31, wid = threadIdx.x >> 5;
  const int row0 = blockIdx.x * 128 + wid * 16;
  const int m = lane & 15, kh = lane >> 4, n = m;
  const int arow = min(row0 + m, N - 1);
  const float dval = deg[arow];
  const float rdeg = 1.0f / fmaxf(dval, 1.0f);
  const bool hasn = dval > 0.0f;

  v8f acc[4] = {};
#pragma unroll
  for (int ks = 0; ks < 6; ++ks) {       // K = 192 = 64(mean)+64(max)+64(root)
    v16h a = load_a_frag(lane, [&](int k) {
      int gk = ks * 32 + k;
      float val;
      if (gk < 64)       val = sum[(size_t)arow * 64 + gk] * rdeg;
      else if (gk < 128) val = hasn ? decf(mx[(size_t)arow * 64 + (gk - 64)]) : 0.0f;
      else               val = h[(size_t)arow * 64 + (gk - 128)];
      return val;
    });
#pragma unroll
    for (int nt = 0; nt < 4; ++nt)
      acc[nt] = WMMA(a, load_b_packed(fWlr, ks * 4 + nt, lane), acc[nt]);
  }
#pragma unroll
  for (int nt = 0; nt < 4; ++nt)
#pragma unroll
    for (int r = 0; r < 8; ++r) {
      int col = nt * 16 + n;
      tile[wid * 16 + 8 * kh + r][col] = acc[nt][r] + bl[col];
    }
  __syncthreads();
  if (threadIdx.x < 128) {
    int row = blockIdx.x * 128 + threadIdx.x;
    if (row < N) {
      float* trow = tile[threadIdx.x];
      float mu = 0.f;
#pragma unroll
      for (int c = 0; c < 64; ++c) mu += trow[c];
      mu *= (1.0f / 64.0f);
      float var = 0.f;
#pragma unroll
      for (int c = 0; c < 64; ++c) { float d2 = trow[c] - mu; var += d2 * d2; }
      var *= (1.0f / 64.0f);
      float rs = rsqrtf(var + LN_EPS);
#pragma unroll
      for (int c = 0; c < 64; ++c) {
        float yv = (trow[c] - mu) * rs * lng[c] + lnb[c];
        h[(size_t)row * 64 + c] = gelu_exact(yv) + h[(size_t)row * 64 + c];
      }
    }
  }
}

// ---- fused query MLP: z=[h_s|h_t|ea] @W1 -> BN -> GELU -> @W2 -> GELU -> @W3 ----
__global__ __launch_bounds__(128) void qmlp_kernel(
    const float* __restrict__ h, int Nn, const int* __restrict__ qi, int Q,
    const float* __restrict__ ea,
    const _Float16* __restrict__ fW1, const float* __restrict__ b1,
    const float* __restrict__ bng, const float* __restrict__ bnb,
    const float* __restrict__ bnm, const float* __restrict__ bnv,
    const _Float16* __restrict__ fW2, const float* __restrict__ b2,
    const float* __restrict__ W3, const float* __restrict__ b3,
    float* __restrict__ out) {
  __shared__ float gbuf[4][16][128];   // [h_src | h_tgt] tiles, filled by TDM gather
  __shared__ float eabuf[4][16][8];    // edge_attr tiles, filled by async copy
  __shared__ float st[4][16][132];
  const int lane = threadIdx.x & 31, wid = threadIdx.x >> 5;
  const int row0 = blockIdx.x * 64 + wid * 16;
  const int m = lane & 15, kh = lane >> 4, n = m;
  const int arow = min(row0 + m, Q - 1);

  // TDM gather of 16 src rows + 16 tgt rows (4 ops x 8 rows, 32-bit index mode).
  {
    const int r16 = lane & 15;
    const int qrow = min(row0 + r16, Q - 1);
    const int idx = (lane < 16) ? qi[qrow] : qi[Q + qrow];
    const unsigned lds0 =
        __builtin_amdgcn_readfirstlane((unsigned)(uintptr_t)&gbuf[wid][0][0]);
#define RL(j) ((unsigned)__builtin_amdgcn_readlane(idx, (j)))
    tdm_gather8(h, lds0,              (unsigned)Nn, RL(0),  RL(1),  RL(2),  RL(3),
                                                    RL(4),  RL(5),  RL(6),  RL(7));
    tdm_gather8(h, lds0 + 4096,       (unsigned)Nn, RL(8),  RL(9),  RL(10), RL(11),
                                                    RL(12), RL(13), RL(14), RL(15));
    tdm_gather8(h, lds0 + 256,        (unsigned)Nn, RL(16), RL(17), RL(18), RL(19),
                                                    RL(20), RL(21), RL(22), RL(23));
    tdm_gather8(h, lds0 + 4096 + 256, (unsigned)Nn, RL(24), RL(25), RL(26), RL(27),
                                                    RL(28), RL(29), RL(30), RL(31));
#undef RL
  }
  // edge_attr tile via async global->LDS (2 lanes per row, 16B each)
  {
    const unsigned lds0 = (unsigned)(uintptr_t)&eabuf[wid][0][0];
    int r = lane >> 1, half = lane & 1;
    int qrow = min(row0 + r, Q - 1);
    async_b128_to_lds(lds0 + (unsigned)(r * 32 + half * 16),
                      (unsigned)(qrow * 32 + half * 16), ea);
  }
  __builtin_amdgcn_s_wait_tensorcnt(0);
  wait_asynccnt0();
  __syncthreads();

  // stage 1: K = 136 padded to 160, N = 128
  v8f acc[8] = {};
#pragma unroll
  for (int ks = 0; ks < 5; ++ks) {
    v16h a = load_a_frag(lane, [&](int k) {
      int gk = ks * 32 + k;
      float val;
      if (gk < 128)      val = gbuf[wid][m][gk];
      else if (gk < 136) val = eabuf[wid][m][gk - 128];
      else               val = 0.0f;
      return val;
    });
#pragma unroll
    for (int nt = 0; nt < 8; ++nt)
      acc[nt] = WMMA(a, load_b_packed(fW1, ks * 8 + nt, lane), acc[nt]);
  }
  // BN (eval) + GELU in-fragment, stash to LDS
#pragma unroll
  for (int nt = 0; nt < 8; ++nt) {
    int col = nt * 16 + n;
    float sc = bng[col] * rsqrtf(bnv[col] + BN_EPS);
    float sh = bnb[col] - bnm[col] * sc;
#pragma unroll
    for (int r = 0; r < 8; ++r) {
      float z = acc[nt][r] + b1[col];
      st[wid][8 * kh + r][col] = gelu_exact(z * sc + sh);
    }
  }
  __syncthreads();
  // stage 2: K = 128 from LDS, N = 64
  v8f acc2[4] = {};
#pragma unroll
  for (int ks = 0; ks < 4; ++ks) {
    v16h a = load_a_frag(lane, [&](int k) { return st[wid][m][ks * 32 + k]; });
#pragma unroll
    for (int nt = 0; nt < 4; ++nt)
      acc2[nt] = WMMA(a, load_b_packed(fW2, ks * 4 + nt, lane), acc2[nt]);
  }
  __syncthreads();
#pragma unroll
  for (int nt = 0; nt < 4; ++nt) {
    int col = nt * 16 + n;
#pragma unroll
    for (int r = 0; r < 8; ++r)
      st[wid][8 * kh + r][col] = gelu_exact(acc2[nt][r] + b2[col]);
  }
  __syncthreads();
  // stage 3: per-row dot with W3[64,1]
  if (threadIdx.x < 64) {
    int row = blockIdx.x * 64 + threadIdx.x;
    if (row < Q) {
      int w = threadIdx.x >> 4, mm = threadIdx.x & 15;
      float dot = b3[0];
#pragma unroll
      for (int c = 0; c < 64; ++c) dot += st[w][mm][c] * W3[c];
      out[row] = dot;
    }
  }
}

extern "C" void kernel_launch(void* const* d_in, const int* in_sizes, int n_in,
                              void* d_out, int out_size, void* d_ws, size_t ws_size,
                              hipStream_t stream) {
  const float* x    = (const float*)d_in[0];
  const int*   ei   = (const int*)d_in[1];
  const float* ea   = (const float*)d_in[2];
  const int*   qi   = (const int*)d_in[3];
  const float* Wenc = (const float*)d_in[4];
  const float* benc = (const float*)d_in[5];
  const float* Wl   = (const float*)d_in[6];
  const float* bl   = (const float*)d_in[7];
  const float* Wr   = (const float*)d_in[8];
  const float* lng  = (const float*)d_in[9];
  const float* lnb  = (const float*)d_in[10];
  const float* W1   = (const float*)d_in[11];
  const float* b1   = (const float*)d_in[12];
  const float* bng  = (const float*)d_in[13];
  const float* bnb  = (const float*)d_in[14];
  const float* bnm  = (const float*)d_in[15];
  const float* bnv  = (const float*)d_in[16];
  const float* W2   = (const float*)d_in[17];
  const float* b2   = (const float*)d_in[18];
  const float* W3   = (const float*)d_in[19];
  const float* b3   = (const float*)d_in[20];

  const int N = in_sizes[0] / 32;   // nodes
  const int E = in_sizes[1] / 2;    // edges
  const int Q = in_sizes[3] / 2;    // query edges

  // ---- workspace layout (all 32B-aligned) ----
  float*    h    = (float*)d_ws;
  float*    sum  = h + (size_t)N * 64;
  unsigned* mx   = (unsigned*)(sum + (size_t)N * 64);
  float*    deg  = (float*)(mx + (size_t)N * 64);
  _Float16* xh   = (_Float16*)(deg + N);               // f16 copy of x
  _Float16* fWenc = xh + (size_t)N * 32;               // 4 frags    = 2048 halves
  _Float16* fWlr  = fWenc + 2048;                      // 3*24 frags = 3*12288
  _Float16* fW1   = fWlr + 3 * 12288;                  // 40 frags   = 20480
  _Float16* fW2   = fW1 + 20480;                       // 16 frags   = 8192

  // ---- one-time per-launch weight pre-swizzle (f32 -> f16 fragment layout) ----
  cvt_f16<<<(N * 32 + 255) / 256, 256, 0, stream>>>(x, xh, N * 32);
  prep_frags<<<(2048 + 255) / 256, 256, 0, stream>>>(Wenc, Wenc, 32, 32, 64, fWenc, 2048);
  for (int l = 0; l < 3; ++l)
    prep_frags<<<(12288 + 255) / 256, 256, 0, stream>>>(
        Wl + (size_t)l * 128 * 64, Wr + (size_t)l * 64 * 64, 128, 192, 64,
        fWlr + (size_t)l * 12288, 12288);
  prep_frags<<<(20480 + 255) / 256, 256, 0, stream>>>(W1, W1, 136, 136, 128, fW1, 20480);
  prep_frags<<<(8192 + 255) / 256, 256, 0, stream>>>(W2, W2, 128, 128, 64, fW2, 8192);

  // degrees (reused across layers)
  fill_f32<<<(N + 255) / 256, 256, 0, stream>>>(deg, 0.0f, N);
  deg_accum<<<(E + 255) / 256, 256, 0, stream>>>(ei, E, deg);

  // node encoder
  encode_kernel<<<(N + 127) / 128, 256, 0, stream>>>(xh, fWenc, benc, h, N);

  // 3 SAGE layers (h updated in place; combine blocks only touch own rows)
  for (int l = 0; l < 3; ++l) {
    fill_f32<<<(N * 64 + 255) / 256, 256, 0, stream>>>(sum, 0.0f, N * 64);
    fill_u32<<<(N * 64 + 255) / 256, 256, 0, stream>>>(mx, ENC_NEG_INF, N * 64);
    scatter_kernel<<<(E + 7) / 8, 256, 0, stream>>>(ei, E, h, sum, mx);
    combine_kernel<<<(N + 127) / 128, 256, 0, stream>>>(
        h, sum, mx, deg, fWlr + (size_t)l * 12288, bl + l * 64,
        lng + l * 64, lnb + l * 64, N);
  }

  // fused edge-predictor MLP (TDM gather + async edge_attr staging)
  qmlp_kernel<<<(Q + 63) / 64, 128, 0, stream>>>(
      h, N, qi, Q, ea, fW1, b1, bng, bnb, bnm, bnv, fW2, b2, W3, b3, (float*)d_out);
}
